// Encoder_7215545057695
// MI455X (gfx1250) — matrix operand order repaired and checked by reading, not compiled, not run on hardware
//
#include <hip/hip_runtime.h>
#include <math.h>

// Problem constants (match reference)
#define S_LEN 2048
#define B_DIM 64
#define E_DIM 256
#define H_DIM 256

typedef __attribute__((ext_vector_type(16))) __bf16 v16bf;
typedef __attribute__((ext_vector_type(8)))  __bf16 v8bf;
typedef __attribute__((ext_vector_type(4)))  __bf16 v4bf;
typedef __attribute__((ext_vector_type(8)))  float  v8f;

// Workspace: [0, 128MB) = x_proj f32 in TRANSPOSED layout [S][H][B] with
// b-chunk swizzle: f32 index = (s*H + n)*B + ((b>>3) ^ (n&7))*8 + (b&7).
// Then W_ih bf16 (128 KB).
#define XPROJ_BYTES ((size_t)S_LEN * B_DIM * H_DIM * 4)

// ---------------------------------------------------------------------------
// Kernel 0: one-shot convert W_ih (f32) -> bf16 in workspace.
// ---------------------------------------------------------------------------
__global__ __launch_bounds__(256)
void wih_to_bf16_kernel(const float* __restrict__ w, __bf16* __restrict__ o)
{
    const int i = (blockIdx.x * 256 + threadIdx.x) * 4;   // 16384 threads * 4
    float4 f = *(const float4*)(w + i);
    v4bf v;
    v[0] = (__bf16)f.x; v[1] = (__bf16)f.y; v[2] = (__bf16)f.z; v[3] = (__bf16)f.w;
    *(v4bf*)(o + i) = v;
}

// ---------------------------------------------------------------------------
// Phase 1: x_proj = sentence @ W_ih + bias  (M=131072, K=256, N=256)
// Each wave: 1 M-tile x 8 N-tiles (8 f32 accumulators). A converted once,
// B read pre-converted bf16 (no cvts on B side). Output stored transposed
// + swizzled so each lane writes two b128 per N-tile.
// ---------------------------------------------------------------------------
__global__ __launch_bounds__(256)
void xproj_gemm_kernel(const float* __restrict__ sent,
                       const __bf16* __restrict__ wihb,
                       const float* __restrict__ bias,
                       float* __restrict__ xproj)
{
    const int lane = threadIdx.x & 31;
    const int wave = threadIdx.x >> 5;
    const int g    = blockIdx.x * 8 + wave;   // 0..16383
    const int mt   = g >> 1;                  // M-tile 0..8191
    const int half = g & 1;                   // N-tiles 0..7 or 8..15
    const int ln   = lane & 15;
    const int hi   = lane >> 4;

    const int m = mt * 16 + ln;
    const float* __restrict__ arow = sent + (size_t)m * E_DIM;

    v8f acc[8];
    #pragma unroll
    for (int j = 0; j < 8; ++j) acc[j] = (v8f){};

    #pragma unroll
    for (int kk = 0; kk < 8; ++kk) {
        const int kb = kk * 32 + hi * 8;
        const float4* ap0 = (const float4*)(arow + kb);
        const float4* ap1 = (const float4*)(arow + kb + 16);
        float4 a0 = ap0[0], a1 = ap0[1], a2 = ap1[0], a3 = ap1[1];
        v16bf a;
        a[0]=(__bf16)a0.x; a[1]=(__bf16)a0.y; a[2]=(__bf16)a0.z; a[3]=(__bf16)a0.w;
        a[4]=(__bf16)a1.x; a[5]=(__bf16)a1.y; a[6]=(__bf16)a1.z; a[7]=(__bf16)a1.w;
        a[8]=(__bf16)a2.x; a[9]=(__bf16)a2.y; a[10]=(__bf16)a2.z; a[11]=(__bf16)a2.w;
        a[12]=(__bf16)a3.x; a[13]=(__bf16)a3.y; a[14]=(__bf16)a3.z; a[15]=(__bf16)a3.w;

        // B cols: lane%16 = N col; elements = K rows kb+s (s<8), kb+8+s (s>=8)
        const __bf16* wrow = wihb + (size_t)(kk * 32 + hi * 8) * H_DIM + half * 128 + ln;
        #pragma unroll
        for (int j = 0; j < 8; ++j) {
            v16bf b;
            #pragma unroll
            for (int s = 0; s < 16; ++s)
                b[s] = wrow[(size_t)(s < 8 ? s : s + 8) * H_DIM + j * 16];
            acc[j] = __builtin_amdgcn_wmma_f32_16x16x32_bf16(
                false, a, false, b, (short)0, acc[j], false, false);
        }
    }

    // Transposed + swizzled store: rows of C are 8 consecutive b within one s.
    const int s_idx = mt >> 2;                  // 4 M-tiles per sequence step
    const int cb    = (mt & 3) * 2 + hi;        // b-chunk 0..7 (b = cb*8 + r)
    #pragma unroll
    for (int j = 0; j < 8; ++j) {
        const int n  = half * 128 + j * 16 + ln;
        const float bn = bias[n];
        float4 f0, f1;
        f0.x = acc[j][0] + bn; f0.y = acc[j][1] + bn;
        f0.z = acc[j][2] + bn; f0.w = acc[j][3] + bn;
        f1.x = acc[j][4] + bn; f1.y = acc[j][5] + bn;
        f1.z = acc[j][6] + bn; f1.w = acc[j][7] + bn;
        float* dst = xproj + ((size_t)(s_idx * H_DIM + n) * B_DIM)
                           + ((cb ^ (n & 7)) << 3);
        *(float4*)dst       = f0;
        *(float4*)(dst + 4) = f1;
    }
}

// ---------------------------------------------------------------------------
// Phase 2: persistent single-WGP scan.
// LDS (exactly 320 KB):
//   [0,       128K) W_hh bf16, XOR-swizzled 32B chunks: (k,n) -> k*512 + ((n>>4)^(k&15))*32
//   [128K,    160K) h buf0 bf16, swizzled 16B chunks:  (m,n) -> m*512 + ((n>>3)^(m&15))*16
//   [160K,    192K) h buf1 bf16 (same swizzle)
//   [192K,    320K) x_proj stage, 2 x 64KB f32 (transposed layout), filled by
//                   global_load_async_to_lds_b128 (ASYNCcnt)
// ---------------------------------------------------------------------------
#define W_OFF   0
#define H0_OFF  131072
#define H1_OFF  163840
#define XP_OFF  196608
#define XP_BUF  65536

__global__ __launch_bounds__(1024)
void rnn_scan_kernel(const float* __restrict__ xproj,
                     const float* __restrict__ h0,
                     const float* __restrict__ Whh,
                     float* __restrict__ out)
{
    extern __shared__ char smem[];
    const int tid  = threadIdx.x;
    const int lane = tid & 31;
    const int wave = tid >> 5;
    const int ln   = lane & 15;
    const int hi   = lane >> 4;

    // Kick off async fill of xp stage buf0 with x_proj[s=0] (64 KB, 64 B/thread)
    {
        uint32_t ldsoff = (uint32_t)(XP_OFF + tid * 64);
        uint32_t goff   = (uint32_t)(tid * 64);
        uint64_t base   = (uint64_t)xproj;
        asm volatile(
            "global_load_async_to_lds_b128 %0, %1, %2 offset:0\n\t"
            "global_load_async_to_lds_b128 %0, %1, %2 offset:16\n\t"
            "global_load_async_to_lds_b128 %0, %1, %2 offset:32\n\t"
            "global_load_async_to_lds_b128 %0, %1, %2 offset:48"
            :: "v"(ldsoff), "v"(goff), "s"(base) : "memory");
    }

    // Stage W_hh -> LDS bf16, swizzled (4096 chunks of 32B; 4 per thread)
    for (int j = tid; j < 256 * 16; j += 1024) {
        const int k = j >> 4, c = j & 15;
        const float4* src = (const float4*)(Whh + (size_t)k * H_DIM + c * 16);
        float4 f0 = src[0], f1 = src[1], f2 = src[2], f3 = src[3];
        v8bf lo, hv;
        lo[0]=(__bf16)f0.x; lo[1]=(__bf16)f0.y; lo[2]=(__bf16)f0.z; lo[3]=(__bf16)f0.w;
        lo[4]=(__bf16)f1.x; lo[5]=(__bf16)f1.y; lo[6]=(__bf16)f1.z; lo[7]=(__bf16)f1.w;
        hv[0]=(__bf16)f2.x; hv[1]=(__bf16)f2.y; hv[2]=(__bf16)f2.z; hv[3]=(__bf16)f2.w;
        hv[4]=(__bf16)f3.x; hv[5]=(__bf16)f3.y; hv[6]=(__bf16)f3.z; hv[7]=(__bf16)f3.w;
        char* dst = smem + W_OFF + k * 512 + (((c ^ (k & 15))) << 5);
        *(v8bf*)dst = lo;
        *(v8bf*)(dst + 16) = hv;
    }
    // Stage h0 -> LDS bf16, swizzled (2048 chunks of 16B; 2 per thread)
    for (int j = tid; j < 64 * 32; j += 1024) {
        const int m = j >> 5, c = j & 31;
        const float4* src = (const float4*)(h0 + (size_t)m * H_DIM + c * 8);
        float4 f0 = src[0], f1 = src[1];
        v8bf v;
        v[0]=(__bf16)f0.x; v[1]=(__bf16)f0.y; v[2]=(__bf16)f0.z; v[3]=(__bf16)f0.w;
        v[4]=(__bf16)f1.x; v[5]=(__bf16)f1.y; v[6]=(__bf16)f1.z; v[7]=(__bf16)f1.w;
        *(v8bf*)(smem + H0_OFF + m * 512 + ((c ^ (m & 15)) << 4)) = v;
    }
    asm volatile("s_wait_asynccnt 0x0" ::: "memory");
    __syncthreads();

    // Tile map: 32 waves -> mt = wave>>3 (0..3), nt pair = (wave&7)*2
    const int mt  = wave >> 3;
    const int nt0 = (wave & 7) * 2;
    const int m   = mt * 16 + ln;          // A row; m&15 == ln
    const int cb  = mt * 2 + hi;           // b-chunk for xp reads / out writes

    for (int s = 0; s < S_LEN; ++s) {
        const int  cur  = s & 1;
        char* hcur = smem + (cur ? H1_OFF : H0_OFF);
        char* hnxt = smem + (cur ? H0_OFF : H1_OFF);

        // Async prefetch next step's x_proj slice into the other stage buffer
        if (s + 1 < S_LEN) {
            uint32_t ldsoff = (uint32_t)(XP_OFF + (1 - cur) * XP_BUF + tid * 64);
            uint32_t goff   = (uint32_t)(tid * 64);
            uint64_t base   = (uint64_t)(xproj + (size_t)(s + 1) * B_DIM * H_DIM);
            asm volatile(
                "global_load_async_to_lds_b128 %0, %1, %2 offset:0\n\t"
                "global_load_async_to_lds_b128 %0, %1, %2 offset:16\n\t"
                "global_load_async_to_lds_b128 %0, %1, %2 offset:32\n\t"
                "global_load_async_to_lds_b128 %0, %1, %2 offset:48"
                :: "v"(ldsoff), "v"(goff), "s"(base) : "memory");
        }

        v8f acc[2];
        acc[0] = (v8f){}; acc[1] = (v8f){};

        const char* hrow = hcur + m * 512;
        #pragma unroll
        for (int kk = 0; kk < 8; ++kk) {
            const int c1 = kk * 4 + hi;               // 16B chunk of K=kb..kb+7
            v8bf alo = *(const v8bf*)(hrow + ((c1 ^ ln) << 4));
            v8bf ahi = *(const v8bf*)(hrow + (((c1 + 2) ^ ln) << 4));
            v16bf a = __builtin_shufflevector(alo, ahi,
                0,1,2,3,4,5,6,7,8,9,10,11,12,13,14,15);

            const int kb = kk * 32 + hi * 8;
            #pragma unroll
            for (int t2 = 0; t2 < 2; ++t2) {
                const int nt = nt0 + t2;
                v16bf b;
                #pragma unroll
                for (int s = 0; s < 16; ++s) {
                    const int k = kb + (s < 8 ? s : s + 8);
                    b[s] = *(const __bf16*)(smem + W_OFF + (size_t)k * 512
                                            + ((nt ^ (k & 15)) << 5) + ln * 2);
                }
                acc[t2] = __builtin_amdgcn_wmma_f32_16x16x32_bf16(
                    false, a, false, b, (short)0, acc[t2], false, false);
            }
        }

        // Epilogue: h_new = tanh(acc + xp) ; xp read as 2x b128 per tile
        // from the transposed+swizzled stage buffer.
        const float* xps = (const float*)(smem + XP_OFF + cur * XP_BUF);
        #pragma unroll
        for (int t2 = 0; t2 < 2; ++t2) {
            const int n = (nt0 + t2) * 16 + ln;
            const float4* xv = (const float4*)(xps + (size_t)n * B_DIM
                                               + ((cb ^ (n & 7)) << 3));
            float4 x0 = xv[0], x1 = xv[1];
            float xr[8] = {x0.x, x0.y, x0.z, x0.w, x1.x, x1.y, x1.z, x1.w};
            #pragma unroll
            for (int r = 0; r < 8; ++r) {
                const int mrow = mt * 16 + hi * 8 + r;
                float v = tanhf(acc[t2][r] + xr[r]);
                *(__bf16*)(hnxt + mrow * 512 + (((n >> 3) ^ (mrow & 15)) << 4)
                                 + (n & 7) * 2) = (__bf16)v;
                if (s == S_LEN - 1)
                    out[(size_t)mrow * H_DIM + n] = v;
            }
        }

        asm volatile("s_wait_asynccnt 0x0" ::: "memory"); // next xp slice landed
        __syncthreads();                                  // h_next + xp visible
    }
}

// ---------------------------------------------------------------------------
extern "C" void kernel_launch(void* const* d_in, const int* in_sizes, int n_in,
                              void* d_out, int out_size, void* d_ws, size_t ws_size,
                              hipStream_t stream) {
    const float* sent = (const float*)d_in[0];   // [S,B,E]
    const float* h0   = (const float*)d_in[1];   // [B,H]
    const float* Wih  = (const float*)d_in[2];   // [E,H]
    const float* Whh  = (const float*)d_in[3];   // [H,H]
    const float* bias = (const float*)d_in[4];   // [H]
    float* out    = (float*)d_out;               // [B,H]
    float* xproj  = (float*)d_ws;                // [S][H][B] f32, swizzled
    __bf16* wihb  = (__bf16*)((char*)d_ws + XPROJ_BYTES); // 128 KB bf16 W_ih

    // W_ih f32 -> bf16 (65536 elems, 4/thread)
    wih_to_bf16_kernel<<<64, 256, 0, stream>>>(Wih, wihb);

    // Phase 1: 16384 waves, each 1 M-tile x 8 N-tiles.
    xproj_gemm_kernel<<<2048, 256, 0, stream>>>(sent, wihb, bias, xproj);

    // Phase 2: one persistent workgroup, full 320 KB LDS.
    const int smem_bytes = 327680;
    hipFuncSetAttribute((const void*)rnn_scan_kernel,
                        hipFuncAttributeMaxDynamicSharedMemorySize, smem_bytes);
    rnn_scan_kernel<<<1, 1024, smem_bytes, stream>>>(xproj, h0, Whh, out);
}
